// VQCodebook_51058571214852
// MI455X (gfx1250) — matrix operand order repaired and checked by reading, not compiled, not run on hardware
//
#include <hip/hip_runtime.h>

typedef __attribute__((ext_vector_type(16))) _Float16 v16h;
typedef __attribute__((ext_vector_type(8)))  _Float16 v8h;
typedef __attribute__((ext_vector_type(8)))  float    v8f;
typedef int v4i_gcc __attribute__((vector_size(16)));   // matches async-LDS builtin param

#define NUM_CODE 1024
#define CODE_DIM 256
#define NVEC     32768        // 32 * 32 * 32 flattened spatial vectors
#define CTILES   (NUM_CODE / 16)
#define LDS_ROW_H 264         // 256 halves + 16B pad per staged code row (bank-conflict-free)

// workspace layout (bytes)
#define WS_ZH    0                                   // f16 [NVEC][CODE_DIM]      16 MB
#define WS_CBH   (NVEC * CODE_DIM * 2)               // f16 [NUM_CODE][CODE_DIM]  512 KB
#define WS_CNORM (WS_CBH + NUM_CODE * CODE_DIM * 2)  // f32 [NUM_CODE]            4 KB
#define WS_IDX   (WS_CNORM + NUM_CODE * 4)           // i32 [NVEC]                128 KB

// CDNA5 async global->LDS path (ASYNCcnt), guarded so compile never regresses.
#if defined(__has_builtin)
#  if __has_builtin(__builtin_amdgcn_global_load_async_to_lds_b128)
#    define HAVE_ASYNC_LDS 1
#  endif
#endif
#ifndef HAVE_ASYNC_LDS
#  define HAVE_ASYNC_LDS 0
#endif

__device__ __forceinline__ void wait_async_zero() {
#if HAVE_ASYNC_LDS
#  if __has_builtin(__builtin_amdgcn_s_wait_asynccnt)
    __builtin_amdgcn_s_wait_asynccnt(0);
#  else
    asm volatile("s_wait_asynccnt 0" ::: "memory");
#  endif
#endif
}

// ---------------------------------------------------------------------------
// 1) Codebook: fp32 -> f16 copy + per-code squared L2 norm
// ---------------------------------------------------------------------------
__global__ __launch_bounds__(256)
void vq_prep_codebook(const float* __restrict__ cb,
                      _Float16* __restrict__ cbh,
                      float* __restrict__ cnorm) {
    const int k = blockIdx.x;           // code id
    const int d = threadIdx.x;          // dim
    const float v = cb[k * CODE_DIM + d];
    cbh[k * CODE_DIM + d] = (_Float16)v;

    float s = v * v;
#pragma unroll
    for (int off = 16; off >= 1; off >>= 1)
        s += __shfl_xor(s, off, 32);
    __shared__ float part[8];
    if ((threadIdx.x & 31) == 0) part[threadIdx.x >> 5] = s;
    __syncthreads();
    if (threadIdx.x == 0) {
        float t = 0.f;
#pragma unroll
        for (int i = 0; i < 8; ++i) t += part[i];
        cnorm[k] = t;
    }
}

// ---------------------------------------------------------------------------
// 2) z_e NCHW fp32 -> [N, D] row-major f16 (coalesced writes; reads hit L2)
// ---------------------------------------------------------------------------
__global__ __launch_bounds__(256)
void vq_prep_z(const float* __restrict__ z_e, _Float16* __restrict__ zh) {
    const int tid = blockIdx.x * 256 + threadIdx.x;   // 8,388,608 elements
    const int d   = tid & (CODE_DIM - 1);
    const int n   = tid >> 8;
    const int b   = n >> 10;
    const int hw  = n & 1023;
    zh[tid] = (_Float16)z_e[((size_t)b * CODE_DIM + d) * 1024 + hw];
}

// ---------------------------------------------------------------------------
// 3) WMMA GEMM (scores = zf . cb^T) fused with argmin over codes.
//    One wave per 16-row tile; A resident in VGPRs; B tiles double-buffered
//    in LDS via async global->LDS copies shared by the 8 waves of the block.
// ---------------------------------------------------------------------------
__global__ __launch_bounds__(256)
void vq_gemm_argmin(const _Float16* __restrict__ zh,
                    const _Float16* __restrict__ cbh,
                    const float* __restrict__ cnorm,
                    int* __restrict__ idx_out,
                    float* __restrict__ idxf_out) {
    __shared__ _Float16 ldsB[2][16 * LDS_ROW_H];   // 2 x 8448 B

    const int lane = threadIdx.x & 31;
    const int wave = threadIdx.x >> 5;          // 0..7
    const int l16  = lane & 15;
    const int half = lane >> 4;                 // 0 or 1
    const int rowTile = blockIdx.x * 8 + wave;  // 0..2047
    const int row0 = rowTile * 16;

    // Stage one 16-code x 256-dim B tile into LDS buffer `buf`.
    auto stage = [&](int ct, int buf) {
        for (int c = threadIdx.x; c < 512; c += 256) {
            const int r   = c >> 5;
            const int col = (c & 31) * 8;
            const _Float16* gsrc = cbh + ((size_t)(ct * 16 + r) * CODE_DIM + col);
            _Float16* ldst = &ldsB[buf][r * LDS_ROW_H + col];
#if HAVE_ASYNC_LDS
            __builtin_amdgcn_global_load_async_to_lds_b128(
                (__attribute__((address_space(1))) v4i_gcc*)gsrc,
                (__attribute__((address_space(3))) v4i_gcc*)ldst, 0, 0);
#else
            *(v8h*)ldst = *(const v8h*)gsrc;
#endif
        }
#if !HAVE_ASYNC_LDS
        if (ct + 1 < CTILES)
            __builtin_prefetch(cbh + (size_t)(ct + 1) * 16 * CODE_DIM + threadIdx.x * 16, 0, 1);
#endif
    };

    // Resident A fragments: 16 rows x 256 K, as 8 chunks of K=32.
    // ISA 16-bit A layout: lanes 0-15 hold K 0..7 / 16..23, lanes 16-31 hold
    // K 8..15 / 24..31, row M = lane % 16.
    const _Float16* arow = zh + (size_t)(row0 + l16) * CODE_DIM;
    v16h a[8];
#pragma unroll
    for (int j = 0; j < 8; ++j) {
        const int kb = 32 * j + 8 * half;
        const v8h lo = *(const v8h*)(arow + kb);
        const v8h hi = *(const v8h*)(arow + kb + 16);
#pragma unroll
        for (int t = 0; t < 8; ++t) { a[j][t] = lo[t]; a[j][8 + t] = hi[t]; }
    }

    float minv[8];
    int   mini[8];
#pragma unroll
    for (int r = 0; r < 8; ++r) { minv[r] = 3.0e38f; mini[r] = 0; }

    stage(0, 0);

    for (int ct = 0; ct < CTILES; ++ct) {
        const int buf = ct & 1;
        // Own async copies for tile `ct` done, then block-wide visibility.
        wait_async_zero();
        __syncthreads();
        // Kick the next tile into the other buffer while we compute this one.
        if (ct + 1 < CTILES) stage(ct + 1, buf ^ 1);

        // B fragments: column N = lane%16, lanes 0-15 hold K 0..7 / 16..23,
        // lanes 16-31 hold K 8..15 / 24..31 (same order as A).
        v16h b[8];
#pragma unroll
        for (int j = 0; j < 8; ++j) {
            const v8h blo = *(const v8h*)(&ldsB[buf][l16 * LDS_ROW_H + 32 * j + 8 * half]);
            const v8h bhi = *(const v8h*)(&ldsB[buf][l16 * LDS_ROW_H + 32 * j + 16 + 8 * half]);
            b[j] = __builtin_shufflevector(blo, bhi, 0,1,2,3,4,5,6,7,8,9,10,11,12,13,14,15);
        }

        v8f acc = {};
#pragma unroll
        for (int j = 0; j < 8; ++j)
            acc = __builtin_amdgcn_wmma_f32_16x16x32_f16(false, a[j], false, b[j],
                                                         (short)0, acc, false, false);

        // dist = ||c||^2 - 2 * <z, c>   (||z||^2 constant per row -> dropped)
        const int   n  = ct * 16 + l16;
        const float cn = cnorm[n];
#pragma unroll
        for (int r = 0; r < 8; ++r) {
            const float dval = fmaf(-2.0f, acc[r], cn);
            if (dval < minv[r]) { minv[r] = dval; mini[r] = n; }  // strict < keeps first idx
        }
    }

    // Reduce over the 16 lanes of each half-wave (codes n%16 spread over lanes).
    // C/D layout: lanes 0-15 -> M = r, lanes 16-31 -> M = 8 + r.
#pragma unroll
    for (int off = 8; off >= 1; off >>= 1) {
#pragma unroll
        for (int r = 0; r < 8; ++r) {
            const float ov = __shfl_xor(minv[r], off, 32);
            const int   oi = __shfl_xor(mini[r], off, 32);
            if (ov < minv[r] || (ov == minv[r] && oi < mini[r])) {
                minv[r] = ov; mini[r] = oi;
            }
        }
    }
    if (l16 == 0) {
#pragma unroll
        for (int r = 0; r < 8; ++r) {
            const int row = row0 + 8 * half + r;
            idx_out[row]  = mini[r];
            idxf_out[row] = (float)mini[r];
        }
    }
}

// ---------------------------------------------------------------------------
// 4) Gather z_q in NCHW from fp32 codebook (coalesced writes; codebook + idx
//    are L2-resident)
// ---------------------------------------------------------------------------
__global__ __launch_bounds__(256)
void vq_gather(const float* __restrict__ cb,
               const int* __restrict__ idx,
               float* __restrict__ zq) {
    const int tid = blockIdx.x * 256 + threadIdx.x;  // 8,388,608 = 32*256*32*32
    const int hw  = tid & 1023;
    const int d   = (tid >> 10) & (CODE_DIM - 1);
    const int b   = tid >> 18;
    const int n   = b * 1024 + hw;
    zq[tid] = cb[(size_t)idx[n] * CODE_DIM + d];
}

extern "C" void kernel_launch(void* const* d_in, const int* in_sizes, int n_in,
                              void* d_out, int out_size, void* d_ws, size_t ws_size,
                              hipStream_t stream) {
    const float* z_e = (const float*)d_in[0];   // [32,256,32,32] fp32
    const float* cb  = (const float*)d_in[1];   // [1024,256] fp32
    float* out = (float*)d_out;

    char* ws = (char*)d_ws;
    _Float16* zh   = (_Float16*)(ws + WS_ZH);
    _Float16* cbh  = (_Float16*)(ws + WS_CBH);
    float*    cnrm = (float*)(ws + WS_CNORM);
    int*      idx  = (int*)(ws + WS_IDX);

    float* zq   = out;                                     // 8,388,608 floats (NCHW)
    float* idxf = out + (size_t)NVEC * CODE_DIM;           // 32,768 indices (as float)

    vq_prep_codebook<<<NUM_CODE, 256, 0, stream>>>(cb, cbh, cnrm);
    vq_prep_z<<<(NVEC * CODE_DIM) / 256, 256, 0, stream>>>(z_e, zh);
    vq_gemm_argmin<<<NVEC / 16 / 8, 256, 0, stream>>>(zh, cbh, cnrm, idx, idxf);
    vq_gather<<<(NVEC * CODE_DIM) / 256, 256, 0, stream>>>(cb, idx, zq);
}
